// QGaLoreLinear_47227460386887
// MI455X (gfx1250) — compile-verified, matrix operand order checked by reading, not run on hardware
//
#include <hip/hip_runtime.h>
#include <stdint.h>

// ---------------------------------------------------------------------------
// QGaLore dequant-GEMM for MI455X (gfx1250, wave32, WMMA bf16)
//   C[M,N] = x[M,K] * dequant(qweight)[N,K]^T + bias[N]
//   M=8192 (4*2048), N=4096, K=4096, group size 256 along K.
// ---------------------------------------------------------------------------

typedef __attribute__((ext_vector_type(16))) __bf16 v16bf;
typedef __attribute__((ext_vector_type(8)))  __bf16 v8bf;
typedef __attribute__((ext_vector_type(8)))  float  v8f;

#define M_TOT 8192
#define N_TOT 4096
#define K_TOT 4096
#define GROUPS_PER_ROW (K_TOT / 256)

#define BM 128
#define BN 128
#define BK 32
#define LDSS 40   // bf16 elements per LDS row: 32 data + 8 pad -> 80B rows (16B aligned, conflict-free)

__global__ __launch_bounds__(256) void qgalore_wmma_bf16(
    const float* __restrict__ x,       // [M,K] fp32
    const int*   __restrict__ qw,      // [N,K] int32 (0..255)
    const float* __restrict__ scales,  // [N*K/256]
    const float* __restrict__ zeros,   // [N*K/256]
    const float* __restrict__ bias,    // [N]
    float*       __restrict__ out)     // [M,N]
{
    __shared__ __bf16 As[BM * LDSS];
    __shared__ __bf16 Bs[BN * LDSS];

    const int tid  = threadIdx.x;
    const int lane = tid & 31;
    const int wave = tid >> 5;
    const int wm   = wave >> 1;   // 0..3 : 32-row M slice
    const int wn   = wave & 1;    // 0..1 : 64-col N slice

    const int bn = blockIdx.x * BN;   // output-feature tile
    const int bm = blockIdx.y * BM;   // activation-row tile

    // global-load coordinates: each thread owns 4 x 16-byte vectors per tile
    const int lrow  = tid >> 3;        // 0..31
    const int lcol4 = (tid & 7) * 4;   // fp32/int32 column within 32-wide K tile

    // fragment-load coordinates (ISA 7.12.2, 16-bit operands)
    const int hrow = lane & 15;           // M (for A) / N (for B)
    const int kb   = (lane >> 4) * 8;     // K half select

    v8f acc[2][4];
#pragma unroll
    for (int i = 0; i < 2; ++i)
#pragma unroll
        for (int j = 0; j < 4; ++j)
            acc[i][j] = (v8f){0.f, 0.f, 0.f, 0.f, 0.f, 0.f, 0.f, 0.f};

    for (int kt = 0; kt < K_TOT; kt += BK) {
        // ---- stage next tile into registers (overlaps previous compute) ----
        float4 xa[4];
        int4   qb[4];
        float  sc[4], zp[4];
#pragma unroll
        for (int rr = 0; rr < 4; ++rr) {
            const int row = lrow + rr * 32;
            xa[rr] = *(const float4*)(x + (size_t)(bm + row) * K_TOT + kt + lcol4);
            const int gn = bn + row;
            qb[rr] = *(const int4*)(qw + (size_t)gn * K_TOT + kt + lcol4);
            const int g = gn * GROUPS_PER_ROW + (kt >> 8);
            sc[rr] = scales[g];
            zp[rr] = zeros[g];
        }
        if (kt + BK < K_TOT) {   // global_prefetch_b8 hints for the following tile
            __builtin_prefetch(x  + (size_t)(bm + lrow) * K_TOT + kt + BK + lcol4, 0, 0);
            __builtin_prefetch(qw + (size_t)(bn + lrow) * K_TOT + kt + BK + lcol4, 0, 0);
        }

        __syncthreads();  // previous iteration's LDS reads done

        // ---- convert + store to LDS (8-byte DS stores) ----
#pragma unroll
        for (int rr = 0; rr < 4; ++rr) {
            const int row = lrow + rr * 32;
            union { __bf16 h[4]; uint64_t u; } pa, pb;
            pa.h[0] = (__bf16)xa[rr].x;
            pa.h[1] = (__bf16)xa[rr].y;
            pa.h[2] = (__bf16)xa[rr].z;
            pa.h[3] = (__bf16)xa[rr].w;
            *(uint64_t*)&As[row * LDSS + lcol4] = pa.u;

            pb.h[0] = (__bf16)(((float)qb[rr].x - zp[rr]) * sc[rr]);
            pb.h[1] = (__bf16)(((float)qb[rr].y - zp[rr]) * sc[rr]);
            pb.h[2] = (__bf16)(((float)qb[rr].z - zp[rr]) * sc[rr]);
            pb.h[3] = (__bf16)(((float)qb[rr].w - zp[rr]) * sc[rr]);
            *(uint64_t*)&Bs[row * LDSS + lcol4] = pb.u;
        }

        __syncthreads();  // tile visible to all waves

        // ---- load fragments and issue 8 WMMAs per wave ----
        v16bf a[2], b[4];
#pragma unroll
        for (int i = 0; i < 2; ++i) {
            const __bf16* p = &As[(wm * 32 + i * 16 + hrow) * LDSS + kb];
            v8bf lo = *(const v8bf*)p;          // K = kb .. kb+7
            v8bf hi = *(const v8bf*)(p + 16);   // K = kb+16 .. kb+23
            a[i] = __builtin_shufflevector(lo, hi, 0,1,2,3,4,5,6,7,8,9,10,11,12,13,14,15);
        }
#pragma unroll
        for (int j = 0; j < 4; ++j) {
            const __bf16* p = &Bs[(wn * 64 + j * 16 + hrow) * LDSS + kb];
            v8bf lo = *(const v8bf*)p;
            v8bf hi = *(const v8bf*)(p + 16);
            b[j] = __builtin_shufflevector(lo, hi, 0,1,2,3,4,5,6,7,8,9,10,11,12,13,14,15);
        }
#pragma unroll
        for (int i = 0; i < 2; ++i)
#pragma unroll
            for (int j = 0; j < 4; ++j)
                acc[i][j] = __builtin_amdgcn_wmma_f32_16x16x32_bf16(
                    /*neg_a=*/false, a[i], /*neg_b=*/false, b[j],
                    /*c_mod=*/(short)0, acc[i][j],
                    /*reuse_a=*/false, /*reuse_b=*/false);
    }

    // ---- epilogue: C layout is lane->N, VGPR r -> M = 8*(lane>=16)+r ----
    const int ncol  = lane & 15;
    const int rbase = (lane >> 4) * 8;
#pragma unroll
    for (int j = 0; j < 4; ++j) {
        const int gn = bn + wn * 64 + j * 16 + ncol;
        const float bv = bias[gn];
#pragma unroll
        for (int i = 0; i < 2; ++i) {
#pragma unroll
            for (int r = 0; r < 8; ++r) {
                const int gm = bm + wm * 32 + i * 16 + rbase + r;
                out[(size_t)gm * N_TOT + gn] = acc[i][j][r] + bv;
            }
        }
    }
}

extern "C" void kernel_launch(void* const* d_in, const int* in_sizes, int n_in,
                              void* d_out, int out_size, void* d_ws, size_t ws_size,
                              hipStream_t stream) {
    const float* x      = (const float*)d_in[0];
    const int*   qw     = (const int*)  d_in[1];
    const float* scales = (const float*)d_in[2];
    const float* zeros  = (const float*)d_in[3];
    const float* bias   = (const float*)d_in[4];
    float*       out    = (float*)d_out;

    dim3 grid(N_TOT / BN, M_TOT / BM);   // 32 x 64 = 2048 workgroups
    dim3 block(256);                     // 8 waves (wave32)
    qgalore_wmma_bf16<<<grid, block, 0, stream>>>(x, qw, scales, zeros, bias, out);
}
